// CausalSelfAttention_3521873183643
// MI455X (gfx1250) — matrix operand order repaired and checked by reading, not compiled, not run on hardware
//
#include <hip/hip_runtime.h>
#include <hip/hip_bf16.h>

typedef __attribute__((ext_vector_type(16))) __bf16 v16bf;
typedef __attribute__((ext_vector_type(8)))  __bf16 v8bf;
typedef __attribute__((ext_vector_type(8)))  float  v8f;

#define BB  2
#define TT  2048
#define DD  1024
#define HH  16
#define HDIM 64
#define N3  3072
#define MM  (BB*TT)   // 4096 tokens

// Build a 16-half A operand from two contiguous 8-half chunks (A layout:
// lane l holds row l&15; halfs 0-7 at K=(l>>4)*16, halfs 8-15 at (l>>4)*16+8, contiguous k as B uses).
__device__ __forceinline__ v16bf load_a16(const __bf16* p0, const __bf16* p1) {
  union { v16bf v; v8bf h[2]; } u;
  u.h[0] = *(const v8bf*)p0;
  u.h[1] = *(const v8bf*)p1;
  return u.v;
}

__device__ __forceinline__ v8f wmma_bf16(v16bf a, v16bf b, v8f c) {
  return __builtin_amdgcn_wmma_f32_16x16x32_bf16(false, a, false, b, (short)0, c, false, false);
}

// ---------------- conversion / transpose helpers ----------------

__global__ void k_cvt(const float* __restrict__ in, __bf16* __restrict__ out, int n) {
  int i = blockIdx.x * blockDim.x + threadIdx.x;
  int stride = gridDim.x * blockDim.x;
  for (; i < n; i += stride) out[i] = (__bf16)in[i];
}

// out[n*rows + k] = in[k*cols + n]   (weights -> [N,K] bf16, B-operand friendly)
__global__ void k_transpose(const float* __restrict__ in, __bf16* __restrict__ out,
                            int rows, int cols) {
  int i = blockIdx.x * blockDim.x + threadIdx.x;
  int stride = gridDim.x * blockDim.x;
  int total = rows * cols;
  for (; i < total; i += stride) {
    int n = i / rows;
    int k = i - n * rows;
    out[i] = (__bf16)in[k * cols + n];
  }
}

// ---------------- shared 32x32-tile GEMM core (ping-pong pipelined) ----------------
// Tile = one operand set: A rows [0,16) and [16,32), B cols [0,32).
// Register budget: acc 32 + 2 operand sets 64 + addresses ~ 105 VGPRs.

struct Tile { v16bf a0, a1, b0, b1; };

__device__ __forceinline__ void load_tile(Tile& t,
                                          const __bf16* __restrict__ arow0,
                                          const __bf16* __restrict__ arow1,
                                          const __bf16* __restrict__ bbase,
                                          int kk, int hl) {
  t.a0 = load_a16(arow0 + kk + hl * 16, arow0 + kk + hl * 16 + 8);
  t.a1 = load_a16(arow1 + kk + hl * 16, arow1 + kk + hl * 16 + 8);
  t.b0 = *(const v16bf*)(bbase + kk);
  t.b1 = *(const v16bf*)(bbase + 16 * DD + kk);
}

__device__ __forceinline__ void mma_tile(v8f acc[2][2], const Tile& t) {
  acc[0][0] = wmma_bf16(t.a0, t.b0, acc[0][0]);
  acc[1][0] = wmma_bf16(t.a1, t.b0, acc[1][0]);
  acc[0][1] = wmma_bf16(t.a0, t.b1, acc[0][1]);
  acc[1][1] = wmma_bf16(t.a1, t.b1, acc[1][1]);
}

// acc[2][2] = A[32xDD] @ B^T strip (32 cols); K pipelined in 32-steps, unrolled x2.
__device__ __forceinline__ void gemm_32x32(v8f acc[2][2],
                                           const __bf16* __restrict__ arow0,
                                           const __bf16* __restrict__ arow1,
                                           const __bf16* __restrict__ bbase,
                                           int hl) {
  Tile t0, t1;
  load_tile(t0, arow0, arow1, bbase, 0, hl);
  for (int kk = 0; kk + 64 < DD; kk += 64) {
    load_tile(t1, arow0, arow1, bbase, kk + 32, hl);
    mma_tile(acc, t0);
    load_tile(t0, arow0, arow1, bbase, kk + 64, hl);
    mma_tile(acc, t1);
  }
  load_tile(t1, arow0, arow1, bbase, DD - 32, hl);
  mma_tile(acc, t0);   // k = DD-64
  mma_tile(acc, t1);   // k = DD-32
}

// ---------------- QKV projection: qkv = x @ Wqkv + b, scatter to Q/K/Vt ----------------
// One wave computes a 32x32 output tile. 128 M-tiles x 96 strips = 12288 waves.

__global__ void __launch_bounds__(256) k_qkv(const __bf16* __restrict__ xb,
                                             const __bf16* __restrict__ wt,   // [3072,1024]
                                             const float*  __restrict__ bias,
                                             __bf16* __restrict__ Qb,   // [B,H,T,HD], pre-scaled
                                             __bf16* __restrict__ Kb,   // [B,H,T,HD]
                                             __bf16* __restrict__ Vtb)  // [B,H,HD,T]
{
  int wid  = blockIdx.x * (blockDim.x >> 5) + (threadIdx.x >> 5);
  int lane = threadIdx.x & 31;
  int ll = lane & 15, hl = lane >> 4;
  int tm    = wid & 127;   // 32-row M tile
  int strip = wid >> 7;    // 32-col strip (96 total)
  const __bf16* arow0 = xb + (tm * 32 + ll) * DD;
  const __bf16* arow1 = arow0 + 16 * DD;
  const __bf16* bbase = wt + (strip * 32 + ll) * DD + hl * 16;

  v8f acc[2][2] = {};
  gemm_32x32(acc, arow0, arow1, bbase, hl);

#pragma unroll
  for (int j = 0; j < 2; ++j) {
    int n = strip * 32 + j * 16 + ll;
    float bj = bias[n];
    int region = n >> 10;        // 0=Q 1=K 2=V
    int dm = n & 1023;
    int hh = dm >> 6, hd = dm & 63;
#pragma unroll
    for (int mi = 0; mi < 2; ++mi)
#pragma unroll
      for (int v = 0; v < 8; ++v) {
        int mg = tm * 32 + mi * 16 + v + 8 * hl;   // token row (C layout)
        int b_ = mg >> 11;                         // / T
        int t_ = mg & (TT - 1);
        float val = acc[mi][j][v] + bj;
        int idx = ((b_ * HH + hh) * TT + t_) * HDIM + hd;
        if (region == 0)      Qb[idx] = (__bf16)(val * 0.03125f);  // 1/sqrt(1024)
        else if (region == 1) Kb[idx] = (__bf16)val;
        else                  Vtb[((b_ * HH + hh) * HDIM + hd) * TT + t_] = (__bf16)val;
      }
  }
}

// ---------------- Flash attention: one wave per (b*h, 16-query tile) ----------------

__global__ void __launch_bounds__(128) k_attn(const __bf16* __restrict__ Q,
                                              const __bf16* __restrict__ K,
                                              const __bf16* __restrict__ Vt,
                                              __bf16* __restrict__ Y)   // [B,T,D] bf16
{
  __shared__ __bf16 ptile[4][16 * 32];  // per-wave 16x32 P tile
  int wslot = threadIdx.x >> 5;
  int wid   = blockIdx.x * 4 + wslot;
  int lane  = threadIdx.x & 31;
  int ll = lane & 15, hl = lane >> 4;
  int bh = wid >> 7;        // 128 q-tiles per (b,h)
  int qt = wid & 127;

  const __bf16* qp = Q  + bh * TT * HDIM;
  const __bf16* kp = K  + bh * TT * HDIM;
  const __bf16* vp = Vt + bh * HDIM * TT;
  __bf16* lds = &ptile[wslot][0];

  const __bf16* qrow = qp + (qt * 16 + ll) * HDIM;
  v16bf aq0 = load_a16(qrow + hl * 16,      qrow + hl * 16 + 8);   // K 0..31
  v16bf aq1 = load_a16(qrow + 32 + hl * 16, qrow + 40 + hl * 16);   // K 32..63

  auto loadK = [&](int kb, v16bf& c00, v16bf& c01, v16bf& c10, v16bf& c11) {
    const __bf16* kr0 = kp + (kb + ll) * HDIM + hl * 16;
    c00 = *(const v16bf*)kr0;
    c01 = *(const v16bf*)(kr0 + 32);
    const __bf16* kr1 = kp + (kb + 16 + ll) * HDIM + hl * 16;
    c10 = *(const v16bf*)kr1;
    c11 = *(const v16bf*)(kr1 + 32);
  };
  auto loadV = [&](int kb, v16bf* bv) {
#pragma unroll
    for (int dt = 0; dt < 4; ++dt)
      bv[dt] = *(const v16bf*)(vp + (dt * 16 + ll) * TT + kb + hl * 16);
  };

  float mx[8], lsum[8];
#pragma unroll
  for (int v = 0; v < 8; ++v) { mx[v] = -__builtin_inff(); lsum[v] = 0.f; }
  v8f o[4] = {};

  int qlast = qt * 16 + 15;
  v16bf c00, c01, c10, c11, bv[4];
  loadK(0, c00, c01, c10, c11);
  loadV(0, bv);

  for (int kb = 0; kb <= qlast; kb += 32) {
    // S = Q @ K^T for 32 keys -> two 16x16 C tiles
    v8f s0, s1;
    {
      v8f c = {};
      c  = wmma_bf16(aq0, c00, c);
      s0 = wmma_bf16(aq1, c01, c);
    }
    {
      v8f c = {};
      c  = wmma_bf16(aq0, c10, c);
      s1 = wmma_bf16(aq1, c11, c);
    }

    // prefetch next key block behind the softmax VALU work
    bool more = (kb + 32) <= qlast;
    v16bf n00, n01, n10, n11, nbv[4];
    if (more) {
      loadK(kb + 32, n00, n01, n10, n11);
      loadV(kb + 32, nbv);
    }

    int k0 = kb + ll, k1 = kb + 16 + ll;
    float alpha[8];
#pragma unroll
    for (int v = 0; v < 8; ++v) {
      int qg = qt * 16 + v + 8 * hl;
      float e0 = (k0 <= qg) ? s0[v] : -__builtin_inff();
      float e1 = (k1 <= qg) ? s1[v] : -__builtin_inff();
      // per-row max across the 16 lanes holding this row
      float mloc = fmaxf(e0, e1);
      mloc = fmaxf(mloc, __shfl_xor(mloc, 1, 32));
      mloc = fmaxf(mloc, __shfl_xor(mloc, 2, 32));
      mloc = fmaxf(mloc, __shfl_xor(mloc, 4, 32));
      mloc = fmaxf(mloc, __shfl_xor(mloc, 8, 32));
      float mnew = fmaxf(mx[v], mloc);
      float p0 = __expf(e0 - mnew);   // exp(-inf)=0 handles mask
      float p1 = __expf(e1 - mnew);
      float rs = p0 + p1;
      rs += __shfl_xor(rs, 1, 32);
      rs += __shfl_xor(rs, 2, 32);
      rs += __shfl_xor(rs, 4, 32);
      rs += __shfl_xor(rs, 8, 32);
      float a_ = __expf(mx[v] - mnew);
      lsum[v] = lsum[v] * a_ + rs;
      mx[v] = mnew;
      alpha[v] = a_;
      // park P tile in LDS (C layout -> memory [m][n])
      int mrow = v + 8 * hl;
      lds[mrow * 32 + ll]      = (__bf16)p0;
      lds[mrow * 32 + 16 + ll] = (__bf16)p1;
    }
#pragma unroll
    for (int dt = 0; dt < 4; ++dt)
#pragma unroll
      for (int v = 0; v < 8; ++v) o[dt][v] *= alpha[v];

    // re-read P in A-operand layout (same-wave DS ops are in-order)
    const __bf16* pr = lds + ll * 32 + hl * 16;
    v16bf ap = load_a16(pr, pr + 8);
#pragma unroll
    for (int dt = 0; dt < 4; ++dt)
      o[dt] = wmma_bf16(ap, bv[dt], o[dt]);

    if (more) {
      c00 = n00; c01 = n01; c10 = n10; c11 = n11;
#pragma unroll
      for (int dt = 0; dt < 4; ++dt) bv[dt] = nbv[dt];
    }
  }

  float inv[8];
#pragma unroll
  for (int v = 0; v < 8; ++v) inv[v] = 1.0f / lsum[v];
  int b_ = bh >> 4, h_ = bh & 15;
#pragma unroll
  for (int dt = 0; dt < 4; ++dt) {
    int d = h_ * HDIM + dt * 16 + ll;
#pragma unroll
    for (int v = 0; v < 8; ++v) {
      int qg = qt * 16 + v + 8 * hl;
      Y[(b_ * TT + qg) * DD + d] = (__bf16)(o[dt][v] * inv[v]);
    }
  }
}

// ---------------- output projection: out = Y @ Wo + b_o (fp32 out) ----------------
// 32x32 tile per wave. 128 M-tiles x 32 strips = 4096 waves.

__global__ void __launch_bounds__(256) k_oproj(const __bf16* __restrict__ yb,
                                               const __bf16* __restrict__ wot, // [1024,1024]
                                               const float*  __restrict__ bias,
                                               float* __restrict__ out)
{
  int wid  = blockIdx.x * (blockDim.x >> 5) + (threadIdx.x >> 5);
  int lane = threadIdx.x & 31;
  int ll = lane & 15, hl = lane >> 4;
  int tm    = wid & 127;
  int strip = wid >> 7;   // 32 strips of 32
  const __bf16* arow0 = yb + (tm * 32 + ll) * DD;
  const __bf16* arow1 = arow0 + 16 * DD;
  const __bf16* bbase = wot + (strip * 32 + ll) * DD + hl * 16;

  v8f acc[2][2] = {};
  gemm_32x32(acc, arow0, arow1, bbase, hl);

#pragma unroll
  for (int j = 0; j < 2; ++j) {
    int n = strip * 32 + j * 16 + ll;
    float bj = bias[n];
#pragma unroll
    for (int mi = 0; mi < 2; ++mi)
#pragma unroll
      for (int v = 0; v < 8; ++v) {
        int mg = tm * 32 + mi * 16 + v + 8 * hl;
        out[mg * DD + n] = acc[mi][j][v] + bj;
      }
  }
}

// ---------------- launcher ----------------

extern "C" void kernel_launch(void* const* d_in, const int* in_sizes, int n_in,
                              void* d_out, int out_size, void* d_ws, size_t ws_size,
                              hipStream_t stream) {
  const float* x    = (const float*)d_in[0];
  const float* wqkv = (const float*)d_in[1];
  const float* bqkv = (const float*)d_in[2];
  const float* wo   = (const float*)d_in[3];
  const float* bo   = (const float*)d_in[4];
  float* out = (float*)d_out;

  char* ws = (char*)d_ws;
  __bf16* xb    = (__bf16*)(ws);                       // 8 MB
  __bf16* wqkvt = (__bf16*)(ws + (size_t)( 8u << 20)); // 6 MB
  __bf16* wot   = (__bf16*)(ws + (size_t)(14u << 20)); // 2 MB
  __bf16* Qb    = (__bf16*)(ws + (size_t)(16u << 20)); // 8 MB
  __bf16* Kb    = (__bf16*)(ws + (size_t)(24u << 20)); // 8 MB
  __bf16* Vtb   = (__bf16*)(ws + (size_t)(32u << 20)); // 8 MB
  __bf16* Yb    = (__bf16*)(ws + (size_t)(40u << 20)); // 8 MB

  k_cvt      <<<2048, 256, 0, stream>>>(x, xb, MM * DD);
  k_transpose<<<2048, 256, 0, stream>>>(wqkv, wqkvt, DD, N3);
  k_transpose<<<1024, 256, 0, stream>>>(wo,   wot,   DD, DD);
  k_qkv      <<<1536, 256, 0, stream>>>(xb, wqkvt, bqkv, Qb, Kb, Vtb); // 12288 waves
  k_attn     <<<1024, 128, 0, stream>>>(Qb, Kb, Vtb, Yb);              // 4096 waves
  k_oproj    <<< 512, 256, 0, stream>>>(Yb, wot, bo, out);             // 4096 waves
}
